// TimesBlock_13889924235673
// MI455X (gfx1250) — matrix operand order repaired and checked by reading, not compiled, not run on hardware
//
#include <hip/hip_runtime.h>
#include <hip/hip_bf16.h>
#include <math.h>

typedef __attribute__((ext_vector_type(16))) _Float16 v16h;
typedef __attribute__((ext_vector_type(8)))  float    v8f;

#define Lq    2048
#define Bq    8
#define Dq    64
#define NFREQ 1025
#define MAXMP 4096            // upper bound on padded rows per batch (n*p < 2*L)
#define MAXPIX (Bq * MAXMP)   // 32768 worst-case output pixels

union FragH { v16h h; uint4 q[2]; };

// ---------------------------------------------------------------------------
// Stage 0a: amplitude spectrum.  freq[j] = mean_{b,d} |X_bd[j]|,
// ampb[j*8+b] = mean_d |X_bd[j]|.  One block per frequency bin; each thread
// handles 2 of the 512 (b,d) series with a rotation-recurrence DFT.
// ---------------------------------------------------------------------------
__global__ void dft_amp_kernel(const float* __restrict__ x,
                               float* __restrict__ freq,
                               float* __restrict__ ampb) {
  const int j = blockIdx.x;
  const int tid = threadIdx.x;
  __shared__ float amps[512];
  __shared__ float bsum[8];
  if (j == 0) {                 // DC is zeroed by the reference
    if (tid == 0) freq[0] = 0.0f;
    if (tid < 8)  ampb[tid] = 0.0f;
    return;
  }
  const float w = -6.283185307179586f * (float)j / (float)Lq;
  const float cs = cosf(w), sn = sinf(w);
  const int p0 = tid, p1 = tid + 256;
  const int b0 = p0 >> 6, d0 = p0 & 63;
  const int b1 = p1 >> 6, d1 = p1 & 63;
  const float* x0 = x + (size_t)b0 * Lq * Dq + d0;
  const float* x1 = x + (size_t)b1 * Lq * Dq + d1;
  float re0 = 0.f, im0 = 0.f, re1 = 0.f, im1 = 0.f;
  for (int tb = 0; tb < Lq; tb += 128) {       // re-seed twiddle every 128 steps
    float c = cosf(w * (float)tb);
    float s = sinf(w * (float)tb);
    for (int t = tb; t < tb + 128; ++t) {
      float v0 = x0[(size_t)t * Dq];
      float v1 = x1[(size_t)t * Dq];
      re0 += v0 * c; im0 += v0 * s;
      re1 += v1 * c; im1 += v1 * s;
      float cn = c * cs - s * sn;
      s = s * cs + c * sn;
      c = cn;
    }
  }
  amps[p0] = sqrtf(re0 * re0 + im0 * im0);
  amps[p1] = sqrtf(re1 * re1 + im1 * im1);
  __syncthreads();
  if (tid < 8) {
    float ssum = 0.f;
    for (int d = 0; d < 64; ++d) ssum += amps[tid * 64 + d];
    ampb[(size_t)j * 8 + tid] = ssum * (1.0f / 64.0f);
    bsum[tid] = ssum;
  }
  __syncthreads();
  if (tid == 0) {
    float tot = 0.f;
    for (int b = 0; b < 8; ++b) tot += bsum[b];
    freq[j] = tot * (1.0f / 512.0f);
  }
}

// ---------------------------------------------------------------------------
// Stage 0b: top-3 bins -> periods + per-batch softmax weights (tiny; serial).
// meta: [0..2]=idx  [3..5]=p  [6..8]=n  [9..11]=n*p
// ---------------------------------------------------------------------------
__global__ void topk_kernel(const float* __restrict__ freq,
                            const float* __restrict__ ampb,
                            int* __restrict__ meta,
                            float* __restrict__ wsoft) {
  if (threadIdx.x != 0 || blockIdx.x != 0) return;
  int idx[3];
  for (int kk = 0; kk < 3; ++kk) {
    float bv = -1.0e30f; int bi = 1;
    for (int jj = 1; jj < NFREQ; ++jj) {
      bool taken = false;
      for (int q = 0; q < kk; ++q) taken |= (idx[q] == jj);
      if (taken) continue;
      float f = freq[jj];
      if (f > bv) { bv = f; bi = jj; }
    }
    idx[kk] = bi;
    int p = Lq / bi;
    int rem = Lq % p;
    int pad = rem ? (p - rem) : 0;
    int nrows = (Lq + pad) / p;
    meta[kk]     = bi;
    meta[3 + kk] = p;
    meta[6 + kk] = nrows;
    meta[9 + kk] = nrows * p;
  }
  for (int b = 0; b < Bq; ++b) {
    float wv[3]; float m = -1.0e30f;
    for (int kk = 0; kk < 3; ++kk) { wv[kk] = ampb[(size_t)idx[kk] * 8 + b]; m = fmaxf(m, wv[kk]); }
    float s = 0.f;
    for (int kk = 0; kk < 3; ++kk) { wv[kk] = expf(wv[kk] - m); s += wv[kk]; }
    for (int kk = 0; kk < 3; ++kk) wsoft[b * 3 + kk] = wv[kk] / s;
  }
}

// ---------------------------------------------------------------------------
// Pack f32 input -> padded f16 activation tensor [B, n*p, 64] (zeros past L).
// ---------------------------------------------------------------------------
__global__ void pack_input_kernel(const float* __restrict__ x,
                                  _Float16* __restrict__ act,
                                  const int* __restrict__ meta, int kp) {
  const int Mp = meta[9 + kp];
  const long tot = (long)Bq * Mp * Dq;
  const long gid = (long)blockIdx.x * blockDim.x + threadIdx.x;
  if (gid >= tot) return;
  const int d = (int)(gid & 63);
  const long pix = gid >> 6;
  const int b = (int)(pix / Mp);
  const int t = (int)(pix % Mp);
  const float v = (t < Lq) ? x[((size_t)b * Lq + t) * Dq + d] : 0.0f;
  act[gid] = (_Float16)v;
}

// ---------------------------------------------------------------------------
// Weight pack: (s,s,Cin,Cout) f32 HWIO -> f16 [Cout][K] with K = (di*s+dj)*Cin+ci
// (contiguous K per output channel -> two aligned b128 loads per B-fragment).
// ---------------------------------------------------------------------------
__global__ void wpack_kernel(const float* __restrict__ w,
                             _Float16* __restrict__ wp,
                             int Ktot, int Cout) {
  const long tot = (long)Ktot * Cout;
  const long gid = (long)blockIdx.x * blockDim.x + threadIdx.x;
  if (gid >= tot) return;
  const int co   = (int)(gid / Ktot);
  const int kidx = (int)(gid % Ktot);
  wp[gid] = (_Float16)w[(size_t)kidx * Cout + co];
}

// ---------------------------------------------------------------------------
// Implicit-GEMM SAME conv via v_wmma_f32_16x16x32_f16, fully templated so the
// tap/channel decomposition is compile-time (no runtime div in the K-loop).
// One wave owns a 16(M) x 64(N) output strip: 4 accumulator tiles sharing one
// A-fragment per 32-K chunk -> 4 independent WMMAs per A-gather.
// Epilogue fuses bias, optional exact GELU, and (final conv) the
// softmax-weighted accumulate into d_out with truncation to L.
// ---------------------------------------------------------------------------
template <int S, int CIN, int COUT, int GELU, int FINAL>
__global__ void __launch_bounds__(128)
conv_wmma_kernel(const _Float16* __restrict__ actIn,
                 const _Float16* __restrict__ Wp,
                 const float* __restrict__ bias,
                 _Float16* __restrict__ actOut,
                 float* __restrict__ outF,
                 const float* __restrict__ wsoft,
                 const int* __restrict__ meta, int kp) {
  constexpr int NT  = 4;               // N tiles per wave (64 output channels)
  constexpr int KT  = S * S * CIN;     // GEMM K
  constexpr int OFF = S / 2;
  const int p    = meta[3 + kp];
  const int nr   = meta[6 + kp];
  const int Mp   = meta[9 + kp];
  const int Mtot = Bq * Mp;
  const int wave = threadIdx.x >> 5;
  const int lane = threadIdx.x & 31;
  const int m0   = (blockIdx.x * 4 + wave) * 16;
  if (m0 >= Mtot) return;                    // wave-uniform exit (EXEC stays full)
  const int n0  = blockIdx.y * (NT * 16);
  const int l15 = lane & 15;
  const int hi  = lane >> 4;                 // K-half select per WMMA layout
  const int row = m0 + l15;
  const bool rowOK = row < Mtot;
  const int b = row / Mp;
  const int t = row - b * Mp;
  const int i = t / p;                       // only runtime division (per thread)
  const int j = t - i * p;

  // per-wave B row pointers: column n0 + nt*16 + l15, K contiguous
  const _Float16* wrow[NT];
#pragma unroll
  for (int nt = 0; nt < NT; ++nt)
    wrow[nt] = Wp + (size_t)(n0 + nt * 16 + l15) * KT + hi * 8;

  v8f acc[NT];
#pragma unroll
  for (int nt = 0; nt < NT; ++nt) acc[nt] = {};

  for (int di = 0; di < S; ++di) {
    const int ii = i + di - OFF;
    const bool iok = (unsigned)ii < (unsigned)nr;
    for (int dj = 0; dj < S; ++dj) {
      const int jj = j + dj - OFF;
      const bool ok = rowOK && iok && ((unsigned)jj < (unsigned)p);
      const _Float16* abase = actIn + ((size_t)(b * nr + ii) * p + jj) * CIN + hi * 8;
      const int kbase = (di * S + dj) * CIN;
#pragma unroll
      for (int cb = 0; cb < CIN; cb += 32) {
        FragH a;
        if (ok) {                                        // two aligned b128 loads
          a.q[0] = *(const uint4*)(abase + cb);
          a.q[1] = *(const uint4*)(abase + cb + 16);
        } else {
          a.q[0] = make_uint4(0u, 0u, 0u, 0u);
          a.q[1] = make_uint4(0u, 0u, 0u, 0u);
        }
        const int k0 = kbase + cb;
#pragma unroll
        for (int nt = 0; nt < NT; ++nt) {                // 4 WMMAs share one A-frag
          FragH bf;
          bf.q[0] = *(const uint4*)(wrow[nt] + k0);
          bf.q[1] = *(const uint4*)(wrow[nt] + k0 + 16);
          acc[nt] = __builtin_amdgcn_wmma_f32_16x16x32_f16(false, a.h, false, bf.h,
                                                           (short)0, acc[nt], false, false);
        }
      }
    }
  }

#pragma unroll
  for (int nt = 0; nt < NT; ++nt) {
    const int col = n0 + nt * 16 + l15;
    const float bs = bias[col];
#pragma unroll
    for (int v = 0; v < 8; ++v) {            // C/D: row = m0 + v + 8*hi, col fixed
      const int r = m0 + hi * 8 + v;
      if (r >= Mtot) continue;
      float val = acc[nt][v] + bs;
      if (GELU) val = 0.5f * val * (1.0f + erff(val * 0.7071067811865475f));
      if (!FINAL) {
        actOut[(size_t)r * COUT + col] = (_Float16)val;
      } else {
        const int bb = r / Mp;
        const int tt = r - bb * Mp;
        if (tt < Lq)
          outF[((size_t)bb * Lq + tt) * Dq + col] += wsoft[bb * 3 + kp] * val;
      }
    }
  }
}

// ---------------------------------------------------------------------------
extern "C" void kernel_launch(void* const* d_in, const int* in_sizes, int n_in,
                              void* d_out, int out_size, void* d_ws, size_t ws_size,
                              hipStream_t stream) {
  (void)in_sizes; (void)n_in; (void)ws_size;
  const float* x = (const float*)d_in[0];
  const float* wsrc[6] = {(const float*)d_in[1], (const float*)d_in[2], (const float*)d_in[3],
                          (const float*)d_in[7], (const float*)d_in[8], (const float*)d_in[9]};
  const float* bias[6] = {(const float*)d_in[4], (const float*)d_in[5], (const float*)d_in[6],
                          (const float*)d_in[10], (const float*)d_in[11], (const float*)d_in[12]};

  char* ws = (char*)d_ws;
  size_t off = 0;
  auto alloc = [&](size_t bytes) {
    char* r = ws + off;
    off = (off + bytes + 255) & ~(size_t)255;
    return r;
  };
  float* freq  = (float*)alloc(NFREQ * 4);
  float* ampb  = (float*)alloc((size_t)NFREQ * 8 * 4);
  int*   meta  = (int*)alloc(16 * 4);
  float* wsoft = (float*)alloc(Bq * 3 * 4);

  static const int Kt[6] = {64, 9 * 128, 25 * 128, 128, 9 * 64, 25 * 64};
  static const int Co[6] = {128, 128, 128, 64, 64, 64};
  _Float16* wp[6];
  for (int c = 0; c < 6; ++c) wp[c] = (_Float16*)alloc((size_t)Kt[c] * Co[c] * 2);
  _Float16* actA = (_Float16*)alloc((size_t)MAXPIX * 128 * 2);
  _Float16* actB = (_Float16*)alloc((size_t)MAXPIX * 128 * 2);

  hipMemsetAsync(d_out, 0, (size_t)out_size * sizeof(float), stream);

  for (int c = 0; c < 6; ++c) {
    const long tot = (long)Kt[c] * Co[c];
    wpack_kernel<<<(int)((tot + 255) / 256), 256, 0, stream>>>(wsrc[c], wp[c], Kt[c], Co[c]);
  }
  dft_amp_kernel<<<NFREQ, 256, 0, stream>>>(x, freq, ampb);
  topk_kernel<<<1, 32, 0, stream>>>(freq, ampb, meta, wsoft);

  float* out = (float*)d_out;
  for (int kp = 0; kp < 3; ++kp) {
    pack_input_kernel<<<(MAXPIX * Dq) / 256, 256, 0, stream>>>(x, actA, meta, kp);
    // grid.x: 512 blocks * 4 waves covers worst-case 2048 M-tiles; grid.y = COUT/64
    conv_wmma_kernel<1, 64, 128, 0, 0><<<dim3(512, 2), 128, 0, stream>>>(
        actA, wp[0], bias[0], actB, out, wsoft, meta, kp);
    conv_wmma_kernel<3, 128, 128, 0, 0><<<dim3(512, 2), 128, 0, stream>>>(
        actB, wp[1], bias[1], actA, out, wsoft, meta, kp);
    conv_wmma_kernel<5, 128, 128, 1, 0><<<dim3(512, 2), 128, 0, stream>>>(
        actA, wp[2], bias[2], actB, out, wsoft, meta, kp);
    conv_wmma_kernel<1, 128, 64, 0, 0><<<dim3(512, 1), 128, 0, stream>>>(
        actB, wp[3], bias[3], actA, out, wsoft, meta, kp);
    conv_wmma_kernel<3, 64, 64, 0, 0><<<dim3(512, 1), 128, 0, stream>>>(
        actA, wp[4], bias[4], actB, out, wsoft, meta, kp);
    conv_wmma_kernel<5, 64, 64, 0, 1><<<dim3(512, 1), 128, 0, stream>>>(
        actB, wp[5], bias[5], actA, out, wsoft, meta, kp);
  }
}
